// TinyRWKV6_34540126994589
// MI455X (gfx1250) — compile-verified, matrix operand order checked by reading, not run on hardware
//
#include <hip/hip_runtime.h>
#include <hip/hip_bf16.h>

// ---------------- problem constants (from reference) ----------------
constexpr int Bk = 2;        // batch
constexpr int Tk = 4096;     // sequence
constexpr int Ck = 64;       // channels
constexpr int Hk = 2;        // heads
constexpr int Kk = 32;       // head size
constexpr int Vk = 32000;    // vocab
constexpr int Lk = 2;        // layers
constexpr int Mk = Bk * Tk;  // 8192 rows
constexpr int CH = 16;       // scan chunk length
constexpr int NCH = Tk / CH; // 256 sequential chunks

typedef __attribute__((ext_vector_type(16))) __bf16 v16bf;
typedef __attribute__((ext_vector_type(8)))  __bf16 v8bf;
typedef __attribute__((ext_vector_type(8)))  float  v8f;
typedef __attribute__((ext_vector_type(4)))  unsigned int v4u;
typedef __attribute__((ext_vector_type(8)))  int v8i;
typedef __attribute__((ext_vector_type(4)))  int v4i;

#if __has_builtin(__builtin_amdgcn_tensor_load_to_lds) && \
    __has_builtin(__builtin_amdgcn_s_wait_tensorcnt)
#define RWKV_TDM 1
#else
#define RWKV_TDM 0
#endif

// Load the 16 bf16 K-values this lane needs for one WMMA operand tile from a
// row-major [rows][32] bf16 matrix. CDNA5 16-bit A/B layout: lanes 0-15 hold
// K = {0..7, 16..23}, lanes 16-31 hold K = {8..15, 24..31} of their row.
__device__ inline v16bf tile16(const __bf16* __restrict__ m, int row, int lane) {
    int koff = (lane < 16) ? 0 : 8;
    const __bf16* p = m + row * 32 + koff;
    v8bf lo = *(const v8bf*)(p);
    v8bf hi = *(const v8bf*)(p + 16);
    v16bf r;
#pragma unroll
    for (int i = 0; i < 8; ++i) { r[i] = lo[i]; r[i + 8] = hi[i]; }
    return r;
}

__device__ inline v8f wmma_bf16(v16bf a, v16bf b, v8f c) {
    return __builtin_amdgcn_wmma_f32_16x16x32_bf16(
        /*neg_a=*/false, a, /*neg_b=*/false, b,
        /*c_mod=*/(short)0, c, /*reuse_a=*/false, /*reuse_b=*/false);
}

#if RWKV_TDM
// Tensor Data Mover: async 2D tile (16 rows x 32 f32, row stride 64 f32) from
// global -> LDS. D# bitfields per CDNA5 ISA.
__device__ inline void tdm_load_tile(unsigned lds_off, const void* gaddr) {
    unsigned long long ga = (unsigned long long)(uintptr_t)gaddr;
    v4u g0;
    g0[0] = 1u;                                       // count=1 (valid user D#)
    g0[1] = lds_off;                                  // lds_addr (bytes)
    g0[2] = (unsigned)ga;                             // global_addr[31:0]
    g0[3] = (unsigned)((ga >> 32) & 0x01FFFFFFu)      // global_addr[56:32]
          | (2u << 30);                               // type=2 ("image")
    v8i g1;
    g1[0] = 0x20000;                                  // data_size=2 -> 4B elems
    g1[1] = (int)(32u << 16);                         // tensor_dim0 = 32
    g1[2] = (int)(16u << 16);                         // tensor_dim1 = 16
    g1[3] = (int)(32u << 16);                         // tile_dim0  = 32
    g1[4] = 16;                                       // tile_dim1  = 16
    g1[5] = 64;                                       // tensor_dim0_stride = 64
    g1[6] = 0;
    g1[7] = 0;
    v4i z4 = {0, 0, 0, 0};
#if defined(__clang_major__) && __clang_major__ >= 23
    v8i z8 = {0, 0, 0, 0, 0, 0, 0, 0};
    __builtin_amdgcn_tensor_load_to_lds(g0, g1, z4, z4, z8, 0);
#else
    __builtin_amdgcn_tensor_load_to_lds(g0, g1, z4, z4, 0);
#endif
}
#endif

// ---------------- prep kernels ----------------

__global__ void rwkv_gather(const int* __restrict__ tok,
                            const float* __restrict__ embed,
                            float* __restrict__ x) {
    int tid = blockIdx.x * blockDim.x + threadIdx.x;   // Mk*Ck threads
    int m = tid >> 6, c = tid & 63;
    x[tid] = embed[(size_t)tok[m] * Ck + c];
}

__global__ void rwkv_cvt_bf16(const float* __restrict__ src,
                              __bf16* __restrict__ dst, int n) {
    int tid = blockIdx.x * blockDim.x + threadIdx.x;
    if (tid < n) dst[tid] = (__bf16)src[tid];
}

// transpose + convert all weights: Wt[l][mat][n][c] = bf16(W[mat][l][c][n])
__global__ void rwkv_prep_w(const float* __restrict__ Wr, const float* __restrict__ Wk,
                            const float* __restrict__ Wv, const float* __restrict__ Ww,
                            const float* __restrict__ Wo, __bf16* __restrict__ Wt) {
    int tid = blockIdx.x * blockDim.x + threadIdx.x;   // Lk*5*64*64 threads
    if (tid >= Lk * 5 * Ck * Ck) return;
    int l   = tid / (5 * Ck * Ck);
    int rem = tid % (5 * Ck * Ck);
    int mat = rem / (Ck * Ck);
    int n   = (rem >> 6) & 63;
    int c   = rem & 63;
    const float* W = (mat == 0) ? Wr : (mat == 1) ? Wk : (mat == 2) ? Wv
                   : (mat == 3) ? Ww : Wo;
    Wt[tid] = (__bf16)W[l * Ck * Ck + c * Ck + n];
}

// ---------------- WMMA projection: r,k,v,logw = x @ {Wr,Wk,Wv,Ww} ----------
// grid.x = Mk/16, block = 128 (4 waves; wave w computes matrix w).
// Wave 3 stores LOG-decay: logw = -exp(x@Ww) so the scan can cumsum decays.
__global__ void rwkv_proj(const __bf16* __restrict__ xb,
                          const __bf16* __restrict__ Wt,   // [4][64][64] (n-major)
                          float* __restrict__ rb, float* __restrict__ kb,
                          float* __restrict__ vb, float* __restrict__ lwb) {
    int wave = threadIdx.x >> 5, lane = threadIdx.x & 31;
    int m0   = blockIdx.x * 16;
    int arow = m0 + (lane & 15);

    v16bf a0 = tile16(xb + arow * Ck, 0, lane);           // K 0..31
    v16bf a1 = tile16(xb + arow * Ck + 32, 0, lane);      // K 32..63

    const __bf16* Wm = Wt + wave * Ck * Ck;
    float* out = (wave == 0) ? rb : (wave == 1) ? kb : (wave == 2) ? vb : lwb;

    int mbase = m0 + ((lane < 16) ? 0 : 8);
#pragma unroll
    for (int nt = 0; nt < 4; ++nt) {
        int n0 = nt * 16;
        const __bf16* wrow = Wm + (n0 + (lane & 15)) * Ck;
        v16bf b0 = tile16(wrow, 0, lane);
        v16bf b1 = tile16(wrow + 32, 0, lane);
        v8f acc = {};
        acc = wmma_bf16(a0, b0, acc);
        acc = wmma_bf16(a1, b1, acc);
        int ncol = n0 + (lane & 15);
#pragma unroll
        for (int vv = 0; vv < 8; ++vv) {
            float val = acc[vv];
            if (wave == 3) val = -__expf(val);            // log w  (w=exp(-exp))
            out[(mbase + vv) * Ck + ncol] = val;
        }
    }
}

// ---------------- chunked RWKV6 scan (WMMA + TDM double-buffer) ------------
// grid = B*H blocks, 32 threads (1 wave). Chunk n=16:
//   O = R~ @ S  +  [tril(R~K~^T) + diag((r.u)k^T)] @ V
//   S^T <- diag_col(exp(cs_n)) S^T + V^T @ K^    (10 WMMAs / chunk)
// Lane = channel c. TDM streams next chunk's r/k/v/logw tiles into LDS while
// the current chunk's WMMAs run; s_wait_tensorcnt 4 retires the current set.
__global__ void rwkv_scan_chunk(const float* __restrict__ rb, const float* __restrict__ kb,
                                const float* __restrict__ vb, const float* __restrict__ lwb,
                                const float* __restrict__ u,  float* __restrict__ ob) {
    __shared__ __align__(16) __bf16 sRt[CH * 32];   // R~      [t][c]
    __shared__ __align__(16) __bf16 sKt[CH * 32];   // K~      [t][c]
    __shared__ __align__(16) __bf16 sKr[CH * 32];   // K raw   [t][c]
    __shared__ __align__(16) __bf16 sRu[CH * 32];   // r*u     [t][c]
    __shared__ __align__(16) __bf16 sAm[CH * 32];   // masked A[t][s], s 16..31 = 0
    __shared__ __align__(16) __bf16 sVT[32 * 32];   // V^T     [c][t], t 16..31 = 0
    __shared__ __align__(16) __bf16 sKh[32 * 32];   // K^^T    [c][t], t 16..31 = 0
    __shared__ __align__(16) __bf16 sST[32 * 32];   // S^T     [n][c]
#if RWKV_TDM
    __shared__ __align__(16) float  sIn[2][4][CH * 32];  // TDM double buffer
#endif

    int b = blockIdx.x / Hk, h = blockIdx.x % Hk;
    int lane = threadIdx.x;
    float uc = u[h * Kk + lane];

    // zero state mirror + pad regions (written once)
#pragma unroll
    for (int i = 0; i < 32; ++i) sST[i * 32 + lane] = (__bf16)0.f;
#pragma unroll
    for (int t = CH; t < 32; ++t) {
        sVT[lane * 32 + t] = (__bf16)0.f;
        sKh[lane * 32 + t] = (__bf16)0.f;
        if (lane < CH) sAm[lane * 32 + t] = (__bf16)0.f;
    }
    __syncthreads();

#if RWKV_TDM
    auto issue_chunk = [&](int chn, int buf) {
        int gbase = (b * Tk + chn * CH) * Ck + h * Kk;
        unsigned lbase = (unsigned)(uintptr_t)(&sIn[buf][0][0]);
        tdm_load_tile(lbase + 0 * 2048, rb  + gbase);
        tdm_load_tile(lbase + 1 * 2048, kb  + gbase);
        tdm_load_tile(lbase + 2 * 2048, vb  + gbase);
        tdm_load_tile(lbase + 3 * 2048, lwb + gbase);
    };
    issue_chunk(0, 0);
#endif

    v8f S00 = {}, S01 = {}, S10 = {}, S11 = {};     // S^T tiles (f32 master)
    int half8 = (lane < 16) ? 0 : 8;
    int l15 = lane & 15;

    for (int ch = 0; ch < NCH; ++ch) {
        int t0 = ch * CH;
        float r[CH], k[CH], v[CH], lw[CH], cst[CH];
        float cs = 0.f;

#if RWKV_TDM
        if (ch + 1 < NCH) {                 // prefetch next chunk (async)
            issue_chunk(ch + 1, (ch + 1) & 1);
            __builtin_amdgcn_s_wait_tensorcnt(4);   // current 4 tiles landed
        } else {
            __builtin_amdgcn_s_wait_tensorcnt(0);
        }
        const float* pin = &sIn[ch & 1][0][0];
#pragma unroll
        for (int t = 0; t < CH; ++t) {
            r[t]  = pin[0 * CH * 32 + t * 32 + lane];
            k[t]  = pin[1 * CH * 32 + t * 32 + lane];
            v[t]  = pin[2 * CH * 32 + t * 32 + lane];
            lw[t] = pin[3 * CH * 32 + t * 32 + lane];
            cs += lw[t];
            cst[t] = cs;
        }
#else
        int base = (b * Tk + t0) * Ck + h * Kk + lane;
#pragma unroll
        for (int t = 0; t < CH; ++t) {
            r[t]  = rb[base + t * Ck];
            k[t]  = kb[base + t * Ck];
            v[t]  = vb[base + t * Ck];
            lw[t] = lwb[base + t * Ck];
            cs += lw[t];
            cst[t] = cs;
        }
#endif
        float pn = __expf(cs);                      // exp(cs_n) per channel

#pragma unroll
        for (int t = 0; t < CH; ++t) {
            float e_in = __expf(-cst[t]);           // exp(-cs_t)
            float e_ex = __expf(cst[t] - lw[t]);    // exp(cs_{t-1})
            float kt = k[t] * e_in;
            sRt[t * 32 + lane] = (__bf16)(r[t] * e_ex);
            sKt[t * 32 + lane] = (__bf16)kt;
            sKh[lane * 32 + t] = (__bf16)(pn * kt); // k^ = exp(cs_n - cs_t) k
            sKr[t * 32 + lane] = (__bf16)k[t];
            sRu[t * 32 + lane] = (__bf16)(r[t] * uc);
            sVT[lane * 32 + t] = (__bf16)v[t];
        }
        __syncthreads();

        // ---- inter-chunk: O += R~ @ S0 (S^T rows = S columns) ----
        v16bf aR  = tile16(sRt, l15, lane);
        v16bf bS0 = tile16(sST, l15, lane);
        v16bf bS1 = tile16(sST, 16 + l15, lane);
        v8f O0 = {}, O1 = {};
        O0 = wmma_bf16(aR, bS0, O0);
        O1 = wmma_bf16(aR, bS1, O1);

        // ---- intra-chunk attention matrix ----
        v16bf bKt = tile16(sKt, l15, lane);
        v8f Am = {};
        Am = wmma_bf16(aR, bKt, Am);                // R~ @ K~^T
        v16bf aU  = tile16(sRu, l15, lane);
        v16bf bKr = tile16(sKr, l15, lane);
        v8f Bo = {};
        Bo = wmma_bf16(aU, bKr, Bo);                // (r.u) @ k^T -> diag bonus

#pragma unroll
        for (int vv = 0; vv < 8; ++vv) {            // causal mask + u-diagonal
            int t = vv + half8, s = l15;
            float av = (s < t) ? Am[vv] : ((s == t) ? Bo[vv] : 0.f);
            sAm[t * 32 + s] = (__bf16)av;
        }
        __syncthreads();

        // ---- O += A @ V ----
        v16bf aA  = tile16(sAm, l15, lane);
        v16bf bV0 = tile16(sVT, l15, lane);         // also A-op of V^T below
        v16bf bV1 = tile16(sVT, 16 + l15, lane);
        O0 = wmma_bf16(aA, bV0, O0);
        O1 = wmma_bf16(aA, bV1, O1);

        // ---- state: S^T = diag_col(pn) S^T + V^T @ K^ ----
        float pf0 = __shfl(pn, l15, 32);
        float pf1 = __shfl(pn, 16 + l15, 32);
        v16bf bH0 = tile16(sKh, l15, lane);
        v16bf bH1 = tile16(sKh, 16 + l15, lane);
#pragma unroll
        for (int vv = 0; vv < 8; ++vv) {
            S00[vv] *= pf0; S01[vv] *= pf1;
            S10[vv] *= pf0; S11[vv] *= pf1;
        }
        S00 = wmma_bf16(bV0, bH0, S00);
        S01 = wmma_bf16(bV0, bH1, S01);
        S10 = wmma_bf16(bV1, bH0, S10);
        S11 = wmma_bf16(bV1, bH1, S11);

        __syncthreads();                             // sST reads done; refresh
#pragma unroll
        for (int vv = 0; vv < 8; ++vv) {
            int n = vv + half8;
            sST[n * 32 + l15]             = (__bf16)S00[vv];
            sST[n * 32 + 16 + l15]        = (__bf16)S01[vv];
            sST[(16 + n) * 32 + l15]      = (__bf16)S10[vv];
            sST[(16 + n) * 32 + 16 + l15] = (__bf16)S11[vv];
        }

        // ---- store chunk output ----
#pragma unroll
        for (int vv = 0; vv < 8; ++vv) {
            int t = vv + half8;
            int m = (b * Tk + t0 + t) * Ck + h * Kk;
            ob[m + l15]      = O0[vv];
            ob[m + 16 + l15] = O1[vv];
        }
        __syncthreads();
    }
}

// ---------------- out projection + residual: x += o @ Wo ----------------
__global__ void rwkv_outproj(const __bf16* __restrict__ obf,
                             const __bf16* __restrict__ Wot,  // [64][64] n-major
                             float* __restrict__ x) {
    int wave = threadIdx.x >> 5, lane = threadIdx.x & 31;
    int m0   = blockIdx.x * 16;
    int arow = m0 + (lane & 15);

    v16bf a0 = tile16(obf + arow * Ck, 0, lane);
    v16bf a1 = tile16(obf + arow * Ck + 32, 0, lane);

    int n0 = wave * 16;
    const __bf16* wrow = Wot + (n0 + (lane & 15)) * Ck;
    v16bf b0 = tile16(wrow, 0, lane);
    v16bf b1 = tile16(wrow + 32, 0, lane);
    v8f acc = {};
    acc = wmma_bf16(a0, b0, acc);
    acc = wmma_bf16(a1, b1, acc);

    int mbase = m0 + ((lane < 16) ? 0 : 8);
    int ncol  = n0 + (lane & 15);
#pragma unroll
    for (int vv = 0; vv < 8; ++vv)
        x[(mbase + vv) * Ck + ncol] += acc[vv];           // residual
}

// ---------------- logits: out[M,V] = x @ embed^T (NT GEMM, bf16 WMMA) ------
// grid = (Mk/16, Vk/128), block = 256 (8 waves; wave w -> N-subtile w).
// 1.05 GB f32 output => ~45us HBM floor; kernel is a streaming WMMA store.
__global__ void rwkv_logits(const __bf16* __restrict__ xb,
                            const __bf16* __restrict__ eb,
                            float* __restrict__ out) {
    int wave = threadIdx.x >> 5, lane = threadIdx.x & 31;
    int m0 = blockIdx.x * 16;
    int n0 = blockIdx.y * 128 + wave * 16;

    int arow = m0 + (lane & 15);
    v16bf a0 = tile16(xb + arow * Ck, 0, lane);
    v16bf a1 = tile16(xb + arow * Ck + 32, 0, lane);

    __builtin_prefetch(eb + (size_t)(n0 + 128) * Ck, 0, 1);  // global_prefetch_b8

    int brow = n0 + (lane & 15);
    v16bf b0 = tile16(eb + (size_t)brow * Ck, 0, lane);
    v16bf b1 = tile16(eb + (size_t)brow * Ck + 32, 0, lane);

    v8f acc = {};
    acc = wmma_bf16(a0, b0, acc);
    acc = wmma_bf16(a1, b1, acc);

    int mbase = m0 + ((lane < 16) ? 0 : 8);
    int ncol  = n0 + (lane & 15);
#pragma unroll
    for (int vv = 0; vv < 8; ++vv)
        out[(size_t)(mbase + vv) * Vk + ncol] = acc[vv];
}

// ---------------- host-side launch ----------------
extern "C" void kernel_launch(void* const* d_in, const int* in_sizes, int n_in,
                              void* d_out, int out_size, void* d_ws, size_t ws_size,
                              hipStream_t stream) {
    const int*   tok   = (const int*)  d_in[0];
    const float* embed = (const float*)d_in[1];
    const float* Wr    = (const float*)d_in[2];
    const float* Wk    = (const float*)d_in[3];
    const float* Wv    = (const float*)d_in[4];
    const float* Ww    = (const float*)d_in[5];
    const float* Wo    = (const float*)d_in[6];
    const float* u     = (const float*)d_in[7];
    float* out = (float*)d_out;

    char* ws = (char*)d_ws;
    float*  xf   = (float*) (ws + 0);                    // [M,C] activations
    __bf16* xbf  = (__bf16*)(ws + 2097152);
    float*  rb   = (float*) (ws + 3145728);
    float*  kb   = (float*) (ws + 5242880);
    float*  vb   = (float*) (ws + 7340032);
    float*  lwb  = (float*) (ws + 9437184);              // log-decay
    float*  ob   = (float*) (ws + 11534336);
    __bf16* obf  = (__bf16*)(ws + 13631488);
    __bf16* ebf  = (__bf16*)(ws + 14680064);             // [V,C] bf16 embed
    __bf16* Wt   = (__bf16*)(ws + 18776064);             // [L][5][64][64] bf16, n-major

    const int MC = Mk * Ck;          // 524288
    const int VC = Vk * Ck;          // 2048000
    const int NW = Lk * 5 * Ck * Ck; // 40960

    rwkv_prep_w<<<(NW + 255) / 256, 256, 0, stream>>>(Wr, Wk, Wv, Ww, Wo, Wt);
    rwkv_cvt_bf16<<<(VC + 255) / 256, 256, 0, stream>>>(embed, ebf, VC);
    rwkv_gather<<<MC / 256, 256, 0, stream>>>(tok, embed, xf);

    for (int l = 0; l < Lk; ++l) {
        const __bf16* Wtl = Wt + (size_t)l * 5 * Ck * Ck;
        rwkv_cvt_bf16<<<MC / 256, 256, 0, stream>>>(xf, xbf, MC);
        rwkv_proj<<<Mk / 16, 128, 0, stream>>>(xbf, Wtl, rb, kb, vb, lwb);
        rwkv_scan_chunk<<<Bk * Hk, 32, 0, stream>>>(rb, kb, vb, lwb,
                                                    u + l * Hk * Kk, ob);
        rwkv_cvt_bf16<<<MC / 256, 256, 0, stream>>>(ob, obf, MC);
        rwkv_outproj<<<Mk / 16, 128, 0, stream>>>(obf, Wtl + 4 * Ck * Ck, xf);
    }

    rwkv_cvt_bf16<<<MC / 256, 256, 0, stream>>>(xf, xbf, MC);
    dim3 lg(Mk / 16, Vk / 128);
    rwkv_logits<<<lg, 256, 0, stream>>>(xbf, ebf, out);
}
